// MplugOwlVisualAbstractorMultiHeadAttention_1726576854691
// MI455X (gfx1250) — matrix operand order repaired and checked by reading, not compiled, not run on hardware
//
#include <hip/hip_runtime.h>
#include <hip/hip_bf16.h>
#include <math.h>
#include <stdint.h>

#define H 16
#define DH 64
#define DM 1024
#define NQ 64
#define KV 1088
#define BATCH 64

#define TM 128
#define TN 128
#define TK 32
#define LDA 40   // padded LDS stride in bf16 elems (80B, 16B aligned)

typedef __attribute__((ext_vector_type(16))) __bf16 v16bf;
typedef __attribute__((ext_vector_type(8)))  __bf16 v8bf;
typedef __attribute__((ext_vector_type(8)))  float  v8f;
typedef __attribute__((ext_vector_type(4)))  float  v4f;

__device__ __forceinline__ v8f vzero8() {
  v8f z;
#pragma unroll
  for (int i = 0; i < 8; ++i) z[i] = 0.0f;
  return z;
}

__device__ __forceinline__ v16bf combine8(v8bf lo, v8bf hi) {
  v16bf r;
#pragma unroll
  for (int i = 0; i < 8; ++i) { r[i] = lo[i]; r[i + 8] = hi[i]; }
  return r;
}

__device__ __forceinline__ v8f wmma_bf16(v16bf a, v16bf b, v8f c) {
  // D = A(16x32 bf16) * B(32x16 bf16) + C(16x16 f32)
  return __builtin_amdgcn_wmma_f32_16x16x32_bf16(
      /*neg_a=*/false, a, /*neg_b=*/false, b,
      /*c_mod=*/(short)0, c, /*reuse_a=*/false, /*reuse_b=*/false);
}

// CDNA5 async global->LDS copy (ASYNCcnt-tracked), 16B per lane.
// VDST VGPR carries the per-lane LDS byte offset.
__device__ __forceinline__ void async_load_lds_b128(uint32_t lds_off, uint64_t gaddr) {
  asm volatile("global_load_async_to_lds_b128 %0, %1, off"
               :: "v"(lds_off), "v"(gaddr)
               : "memory");
}
__device__ __forceinline__ void wait_asynccnt0() {
  asm volatile("s_wait_asynccnt 0" ::: "memory");
}

// ---------------------------------------------------------------------------
// Stage 0: f32 -> bf16 weight conversion
// ---------------------------------------------------------------------------
__global__ __launch_bounds__(256) void cvt_bf16_kernel(
    const float* __restrict__ s, __bf16* __restrict__ d, int n) {
  int i = blockIdx.x * 256 + threadIdx.x;
  if (i < n) d[i] = (__bf16)s[i];
}

// ---------------------------------------------------------------------------
// Stage 1: projection GEMM  Out = split_heads((X [+ pos]) @ W^T + bias) * scale
//   X: f32 [rows, DM] (rows = BATCH*S), W: bf16 [DM, DM] row-major
//   layoutV==0: Out[b][h][s][dh]   (K, Q)
//   layoutV==1: Out[b][h][dh][s]   (V transposed, for contiguous PV B-frags)
// Tile: 128x128x32, 256 threads = 8 waves (4M x 2N), each wave 32x64.
// A tile: VGPR path (fused pos-add + f32->bf16). B tile: async global->LDS.
// ---------------------------------------------------------------------------
__global__ __launch_bounds__(256) void proj_gemm_kernel(
    const float* __restrict__ X,
    const float* __restrict__ posA, const float* __restrict__ posB, int posSplit,
    const __bf16* __restrict__ W, const float* __restrict__ bias,
    __bf16* __restrict__ Out, int S, int layoutV, float outScale)
{
  __shared__ __align__(16) __bf16 As[TM][LDA];
  __shared__ __align__(16) __bf16 Bs[TN][LDA];

  const int mBase = blockIdx.x * TM;
  const int nBase = blockIdx.y * TN;
  const int tid  = threadIdx.x;
  const int lane = tid & 31;
  const int wave = tid >> 5;
  const int l16  = lane & 15;
  const int hh   = lane >> 4;      // half-wave: 0 or 1
  const int wm   = wave & 3;       // wave M index 0..3
  const int wn   = wave >> 2;      // wave N index 0..1

  v8f acc[2][4];
#pragma unroll
  for (int i = 0; i < 2; ++i)
#pragma unroll
    for (int j = 0; j < 4; ++j) acc[i][j] = vzero8();

  for (int ks = 0; ks < DM; ks += TK) {
    // ---- cooperative tile load: 512 (row, 8-elem chunk) tasks, 2/thread ----
#pragma unroll
    for (int i = 0; i < 2; ++i) {
      int t  = tid + i * 256;
      int r  = t >> 2;
      int k0 = (t & 3) << 3;
      // B tile: bf16 weights -> LDS via CDNA5 async copy (no VGPR round-trip)
      {
        const __bf16* wp = W + (size_t)(nBase + r) * DM + ks + k0;
        uint32_t ldsoff = (uint32_t)(uintptr_t)&Bs[r][k0];
        async_load_lds_b128(ldsoff, (uint64_t)(uintptr_t)wp);
      }
      // A tile: f32 X (+ pos) -> bf16 (needs VALU transform, VGPR path)
      {
        int grow = mBase + r;
        const float* xp = X + (size_t)grow * DM + ks + k0;
        v4f x0 = *(const v4f*)xp;
        v4f x1 = *(const v4f*)(xp + 4);
        if (posA) {
          int s = grow % S;
          const float* pr = (s < posSplit) ? (posA + (size_t)s * DM)
                                           : (posB + (size_t)(s - posSplit) * DM);
          x0 += *(const v4f*)(pr + ks + k0);
          x1 += *(const v4f*)(pr + ks + k0 + 4);
        }
        v8bf xb;
#pragma unroll
        for (int e = 0; e < 4; ++e) { xb[e] = (__bf16)x0[e]; xb[e + 4] = (__bf16)x1[e]; }
        *(v8bf*)&As[r][k0] = xb;
      }
    }
    wait_asynccnt0();     // async LDS writes visible before fragment reads
    __syncthreads();

    // ---- fragments (contiguous k = 16*hh + s order, same in A and B) ----
    v16bf afr[2], bfr[4];
#pragma unroll
    for (int fm = 0; fm < 2; ++fm) {
      int row = wm * 32 + fm * 16 + l16;
      v8bf lo = *(const v8bf*)&As[row][hh * 16];       // k = hh*16 .. +7
      v8bf hi = *(const v8bf*)&As[row][hh * 16 + 8];   // k = hh*16+8 .. +15
      afr[fm] = combine8(lo, hi);
    }
#pragma unroll
    for (int fn = 0; fn < 4; ++fn) {
      int col = wn * 64 + fn * 16 + l16;
      v8bf lo = *(const v8bf*)&Bs[col][hh * 16];       // k = hh*16 .. +7
      v8bf hi = *(const v8bf*)&Bs[col][hh * 16 + 8];   // k = hh*16+8 .. +15
      bfr[fn] = combine8(lo, hi);
    }
#pragma unroll
    for (int fm = 0; fm < 2; ++fm)
#pragma unroll
      for (int fn = 0; fn < 4; ++fn)
        acc[fm][fn] = wmma_bf16(afr[fm], bfr[fn], acc[fm][fn]);
    __syncthreads();
  }

  // ---- epilogue: bias, scale, head-split scatter (D layout: row = r + 8*hh) ----
#pragma unroll
  for (int fm = 0; fm < 2; ++fm) {
#pragma unroll
    for (int fn = 0; fn < 4; ++fn) {
      int gc   = nBase + wn * 64 + fn * 16 + l16;
      float bv = bias[gc];
      int hcol = gc >> 6;          // head
      int dh   = gc & 63;          // dim within head
      int gr0  = mBase + wm * 32 + fm * 16 + hh * 8;
#pragma unroll
      for (int r = 0; r < 8; ++r) {
        int grow = gr0 + r;
        int bidx = grow / S;
        int s    = grow - bidx * S;
        float val = (acc[fm][fn][r] + bv) * outScale;
        size_t oidx;
        if (layoutV)
          oidx = (((size_t)bidx * H + hcol) * DH + dh) * (size_t)S + s;
        else
          oidx = (((size_t)bidx * H + hcol) * (size_t)S + s) * DH + dh;
        Out[oidx] = (__bf16)val;
      }
    }
  }
}

// ---------------------------------------------------------------------------
// Stage 2: flash attention per (b,h). 1024 blocks x 128 threads (4 waves).
// Each wave owns 16 query rows; KV streamed in steps of 32.
//   Q:  bf16 [B,H,NQ,DH]  (pre-scaled by 1/8)
//   K:  bf16 [B,H,KV,DH]
//   Vt: bf16 [B,H,DH,KV]
//   Out: f32 [B,NQ,DM]
// ---------------------------------------------------------------------------
__global__ __launch_bounds__(128) void attn_kernel(
    const __bf16* __restrict__ Q, const __bf16* __restrict__ K,
    const __bf16* __restrict__ Vt, float* __restrict__ Out)
{
  const int bh   = blockIdx.x;     // b*H + h
  const int b    = bh >> 4;
  const int h    = bh & 15;
  const int wave = threadIdx.x >> 5;
  const int lane = threadIdx.x & 31;
  const int l16  = lane & 15;
  const int hh   = lane >> 4;

  __shared__ __align__(16) __bf16 Plds[4][16][LDA];   // per-wave 16x32 P tile

  // Q A-fragments for d = 0..63 (2 frags of 16x32), resident in registers
  const __bf16* qp = Q + ((size_t)bh * NQ + wave * 16 + l16) * DH;
  v16bf aq[2];
#pragma unroll
  for (int f = 0; f < 2; ++f) {
    v8bf lo = *(const v8bf*)(qp + f * 32 + hh * 16);
    v8bf hi = *(const v8bf*)(qp + f * 32 + hh * 16 + 8);
    aq[f] = combine8(lo, hi);
  }

  float m[8], l[8];
  v8f o[4];
#pragma unroll
  for (int r = 0; r < 8; ++r) { m[r] = -INFINITY; l[r] = 0.0f; }
#pragma unroll
  for (int dt = 0; dt < 4; ++dt) o[dt] = vzero8();

  const __bf16* kbase = K  + (size_t)bh * KV * DH;
  const __bf16* vbase = Vt + (size_t)bh * DH * KV;

  for (int kv0 = 0; kv0 < KV; kv0 += 32) {
    // ---- S = Q*K^T for 32 kv columns (two 16x16 tiles) ----
    v8f s_lo = vzero8(), s_hi = vzero8();
#pragma unroll
    for (int f = 0; f < 2; ++f) {
      const __bf16* kp0 = kbase + (size_t)(kv0 + l16) * DH + f * 32 + hh * 16;
      v16bf bk0 = combine8(*(const v8bf*)kp0, *(const v8bf*)(kp0 + 8));
      s_lo = wmma_bf16(aq[f], bk0, s_lo);
      const __bf16* kp1 = kbase + (size_t)(kv0 + 16 + l16) * DH + f * 32 + hh * 16;
      v16bf bk1 = combine8(*(const v8bf*)kp1, *(const v8bf*)(kp1 + 8));
      s_hi = wmma_bf16(aq[f], bk1, s_hi);
    }

    // ---- online softmax; row for accumulator index r is (r + 8*hh) ----
    float pl[8], ph[8], alpha[8];
#pragma unroll
    for (int r = 0; r < 8; ++r) {
      float tm = fmaxf(s_lo[r], s_hi[r]);
#pragma unroll
      for (int off = 8; off >= 1; off >>= 1)
        tm = fmaxf(tm, __shfl_xor(tm, off, 16));
      float mn = fmaxf(m[r], tm);
      alpha[r] = __expf(m[r] - mn);
      pl[r] = __expf(s_lo[r] - mn);
      ph[r] = __expf(s_hi[r] - mn);
      float ts = pl[r] + ph[r];
#pragma unroll
      for (int off = 8; off >= 1; off >>= 1)
        ts += __shfl_xor(ts, off, 16);
      l[r] = l[r] * alpha[r] + ts;
      m[r] = mn;
    }
#pragma unroll
    for (int dt = 0; dt < 4; ++dt)
#pragma unroll
      for (int r = 0; r < 8; ++r)
        o[dt][r] *= alpha[r];

    // ---- P -> LDS (D layout) -> reread as A-frag layout ----
#pragma unroll
    for (int r = 0; r < 8; ++r) {
      Plds[wave][r + 8 * hh][l16]      = (__bf16)pl[r];
      Plds[wave][r + 8 * hh][16 + l16] = (__bf16)ph[r];
    }
    __syncthreads();
    v8bf plo = *(const v8bf*)&Plds[wave][l16][hh * 16];
    v8bf phi = *(const v8bf*)&Plds[wave][l16][hh * 16 + 8];
    v16bf ap = combine8(plo, phi);

    // ---- O += P*V : Vt rows give contiguous 32B B-fragments ----
#pragma unroll
    for (int dt = 0; dt < 4; ++dt) {
      const __bf16* vp = vbase + (size_t)(dt * 16 + l16) * KV + kv0 + hh * 16;
      v16bf bv = combine8(*(const v8bf*)vp, *(const v8bf*)(vp + 8));
      o[dt] = wmma_bf16(ap, bv, o[dt]);
    }
    __syncthreads();
  }

  // ---- normalize and write merged-head output [B, NQ, DM] ----
#pragma unroll
  for (int r = 0; r < 8; ++r) {
    int q = wave * 16 + r + 8 * hh;
    float inv = 1.0f / l[r];
#pragma unroll
    for (int dt = 0; dt < 4; ++dt) {
      int col = h * DH + dt * 16 + l16;
      Out[((size_t)b * NQ + q) * DM + col] = o[dt][r] * inv;
    }
  }
}

// ---------------------------------------------------------------------------
extern "C" void kernel_launch(void* const* d_in, const int* in_sizes, int n_in,
                              void* d_out, int out_size, void* d_ws, size_t ws_size,
                              hipStream_t stream) {
  const float* hs   = (const float*)d_in[0];
  const float* enc  = (const float*)d_in[1];
  const float* Wq   = (const float*)d_in[2];
  const float* bq   = (const float*)d_in[3];
  const float* Wk   = (const float*)d_in[4];
  const float* bk   = (const float*)d_in[5];
  const float* Wv   = (const float*)d_in[6];
  const float* bv   = (const float*)d_in[7];
  const float* qpos = (const float*)d_in[8];
  const float* kpos = (const float*)d_in[9];
  float* out = (float*)d_out;

  char* ws = (char*)d_ws;
  __bf16* Wq_b = (__bf16*)(ws + 0);                          // 2 MB
  __bf16* Wk_b = (__bf16*)(ws + (size_t)(2u << 20));         // 2 MB
  __bf16* Wv_b = (__bf16*)(ws + (size_t)(4u << 20));         // 2 MB
  __bf16* Qb   = (__bf16*)(ws + (size_t)(6u << 20));         // 8 MB
  __bf16* Kb   = (__bf16*)(ws + (size_t)(14u << 20));        // 136 MB
  __bf16* Vtb  = (__bf16*)(ws + (size_t)(14u << 20) + 142606336ull);

  const int nW = DM * DM;
  dim3 cgrid((nW + 255) / 256);
  cvt_bf16_kernel<<<cgrid, 256, 0, stream>>>(Wq, Wq_b, nW);
  cvt_bf16_kernel<<<cgrid, 256, 0, stream>>>(Wk, Wk_b, nW);
  cvt_bf16_kernel<<<cgrid, 256, 0, stream>>>(Wv, Wv_b, nW);

  // K = split_heads((enc + concat(q_pos,k_pos)) @ Wk^T + bk)
  {
    dim3 grid((BATCH * KV) / TM, DM / TN);
    proj_gemm_kernel<<<grid, 256, 0, stream>>>(enc, qpos, kpos, NQ, Wk_b, bk,
                                               Kb, KV, 0, 1.0f);
  }
  // V^T = split_heads(enc @ Wv^T + bv), stored [B,H,DH,KV]
  {
    dim3 grid((BATCH * KV) / TM, DM / TN);
    proj_gemm_kernel<<<grid, 256, 0, stream>>>(enc, nullptr, nullptr, 0, Wv_b, bv,
                                               Vtb, KV, 1, 1.0f);
  }
  // Q = split_heads((hs + q_pos) @ Wq^T + bq) * (1/sqrt(64))
  {
    dim3 grid((BATCH * NQ) / TM, DM / TN);
    proj_gemm_kernel<<<grid, 256, 0, stream>>>(hs, qpos, qpos, NQ, Wq_b, bq,
                                               Qb, NQ, 0, 0.125f);
  }
  attn_kernel<<<BATCH * H, 128, 0, stream>>>(Qb, Kb, Vtb, out);
}
